// LSTMCell_21818433864264
// MI455X (gfx1250) — matrix operand order repaired and checked by reading, not compiled, not run on hardware
//
#include <hip/hip_runtime.h>

// ---- vector types matching CDNA5 WMMA operand shapes ----
typedef _Float16 f16x4  __attribute__((ext_vector_type(4)));
typedef _Float16 f16x8  __attribute__((ext_vector_type(8)));
typedef _Float16 f16x16 __attribute__((ext_vector_type(16)));
typedef float    f32x8  __attribute__((ext_vector_type(8)));

#define BATCH 8192
#define INCH  2048
#define OUTCH 2048
#define GATES 4
#define WROW  (GATES * OUTCH)   // 8192, row stride of W

// GEMM tiling
#define BM 128                  // block rows
#define BN 64                   // block cols per gate
#define KB 32                   // K per WMMA step
#define LDP 40                  // padded LDS row (halves): 80B, 16B-multiple
#define BUFH (BM * LDP + GATES * BN * LDP)  // halves per LDS buffer = 15360

// ---------------- f32 -> f16 conversion of A (x4 vectorized) ----------------
__global__ void lstm_cvt_f32_f16(const float* __restrict__ in,
                                 _Float16* __restrict__ out, int n4) {
  int i = blockIdx.x * blockDim.x + threadIdx.x;
  if (i < n4) {
    float4 v = ((const float4*)in)[i];
    f16x4 h;
    h[0] = (_Float16)v.x; h[1] = (_Float16)v.y;
    h[2] = (_Float16)v.z; h[3] = (_Float16)v.w;
    ((f16x4*)out)[i] = h;
  }
}

// ------- one-time W f32->f16 + transpose: Wt[n][k] = (f16)W[k][n] -------
__global__ void lstm_cvt_w_transpose(const float* __restrict__ W,
                                     _Float16* __restrict__ Wt) {
  __shared__ _Float16 t[32][33];
  const int tx = threadIdx.x;           // 0..31
  const int ty = threadIdx.y;           // 0..7
  const int n0 = blockIdx.x * 32;
  const int k0 = blockIdx.y * 32;
#pragma unroll
  for (int j = 0; j < 32; j += 8)
    t[ty + j][tx] = (_Float16)W[(size_t)(k0 + ty + j) * WROW + n0 + tx];
  __syncthreads();
#pragma unroll
  for (int j = 0; j < 32; j += 8)
    Wt[(size_t)(n0 + ty + j) * INCH + k0 + tx] = t[tx][ty + j];
}

__device__ __forceinline__ float fast_sigmoid(float x) {
  return 1.0f / (1.0f + __expf(-x));
}
__device__ __forceinline__ float fast_tanh(float x) {
  float e = __expf(2.0f * x);
  return 1.0f - 2.0f / (e + 1.0f);
}

// ---- CDNA5 async global->LDS copy (ASYNCcnt path), SADDR+voffset form ----
// The LDS destination operand is derived from a REAL pointer into the shared
// array (ptrtoint => the LDS global escapes, so the "memory" clobber is seen
// as a potential writer of it and the subsequent ds_load fragments survive).
// Aperture layout: generic-pointer low 32 bits == LDS byte address.
// ISA adds INST_OFFSET to BOTH the LDS and the global address, so contiguous
// runs on both sides share one (lds_ptr, g_off) pair across offset:0/16/...
__device__ __forceinline__ void async_b128(const _Float16* lds_ptr, int g_off,
                                           const void* base, int ioff) {
  unsigned lds_addr = (unsigned)(uintptr_t)lds_ptr;
  asm volatile("global_load_async_to_lds_b128 %0, %1, %2 offset:%3"
               :: "v"(lds_addr), "v"(g_off), "s"(base), "i"(ioff)
               : "memory");
}
__device__ __forceinline__ void wait_async0() {
  asm volatile("s_wait_asynccnt 0x0" ::: "memory");
}

// ---------------- fused LSTM-cell GEMM (WMMA f16 -> f32) ----------------
// Block: 256 threads = 8 waves in a 4(M) x 2(N) grid.
// Block tile: M=128 x N=64 per gate (x4 gates). Wave tile: 32x32 per gate.
// Per wave: 2(M) x 2(N) x 4(gates) = 16 f32 16x16 accumulators.
// K-slabs staged via async global->LDS copies, ping-pong LDS buffers,
// ONE barrier per K-step.
__global__ __launch_bounds__(256) void lstm_wmma_gemm(
    const _Float16* __restrict__ Ah,   // [BATCH, INCH]   f16 row-major
    const _Float16* __restrict__ Wt,   // [4*OUTCH, INCH] f16 (W transposed)
    const float*    __restrict__ bias, // [4*OUTCH]
    const float*    __restrict__ cell, // [BATCH, OUTCH]
    float*          __restrict__ outH, // [BATCH, OUTCH]
    float*          __restrict__ outC) // [BATCH, OUTCH]
{
  // [buffer p][ A: BM x LDP | W: 256 x LDP ]
  __shared__ _Float16 smem[2 * BUFH];
  // belt-and-braces: escape the LDS array so asm clobbers may alias it
  asm volatile("" :: "v"(smem) : "memory");

  const int tid  = threadIdx.x;
  const int lane = tid & 31;
  const int wave = tid >> 5;
  const int wm   = wave >> 1;                // 0..3 : 32-row slab
  const int wn   = wave & 1;                 // 0..1 : 32-col slab (per gate)
  const int blockM = blockIdx.y * BM;
  const int blockN = blockIdx.x * BN;

  f32x8 acc[16] = {};   // acc[((g*2+ni)*2)+mi]

  // --- async staging: A, each thread copies 16 halves of one row (2x16B) ---
  const int a_row = tid >> 1;                // 0..127
  const int a_col = (tid & 1) * 16;          // 0 or 16
  _Float16* aLds = &smem[a_row * LDP + a_col];
  const int aOff0 = ((blockM + a_row) * INCH + a_col) * 2;     // bytes

  // --- async staging: W, each thread copies one n's 32-half K-run (4x16B) ---
  const int w_g = tid >> 6;                  // gate 0..3
  const int w_c = tid & 63;                  // col within gate tile
  _Float16* wLds = &smem[BM * LDP + tid * LDP];
  const int wOff0 = ((w_g * OUTCH + blockN + w_c) * INCH) * 2; // bytes

  // --- fragment gather offsets (CDNA5 16-bit WMMA layouts) ---
  const int nl   = lane & 15;
  const int a_kb = (lane & 16) ? 8 : 0;      // A: K {0..7,16..23}/{8..15,24..31}
  const int b_kb = (lane & 16) ? 8 : 0;      // B: K {0..7,16..23}/{8..15,24..31}
  const int a_base = (wm * 32 + nl) * LDP + a_kb;
  const int b_base = (wn * 32 + nl) * LDP + b_kb;

  // prologue: issue async copies of the first K slab into buffer 0
  async_b128(aLds, aOff0, Ah, 0);
  async_b128(aLds, aOff0, Ah, 16);
  async_b128(wLds, wOff0, Wt, 0);
  async_b128(wLds, wOff0, Wt, 16);
  async_b128(wLds, wOff0, Wt, 32);
  async_b128(wLds, wOff0, Wt, 48);

  int p = 0;
  for (int kt = 0; kt < INCH; kt += KB) {
    wait_async0();     // this wave's copies for buffer p have landed in LDS
    __syncthreads();   // all waves' copies landed; all reads of p^1 finished

    // issue next slab's async copies into the other buffer; they proceed
    // concurrently with the WMMA stream below
    if (kt + KB < INCH) {
      const int kb = (kt + KB) * 2;        // global byte advance
      const int lo = (p ^ 1) * BUFH;       // other LDS buffer (halves)
      async_b128(aLds + lo, aOff0 + kb, Ah, 0);
      async_b128(aLds + lo, aOff0 + kb, Ah, 16);
      async_b128(wLds + lo, wOff0 + kb, Wt, 0);
      async_b128(wLds + lo, wOff0 + kb, Wt, 16);
      async_b128(wLds + lo, wOff0 + kb, Wt, 32);
      async_b128(wLds + lo, wOff0 + kb, Wt, 48);
    }

    const _Float16* sA = smem + p * BUFH;
    const _Float16* sW = sA + BM * LDP;

    // A fragments: 2 M-subtiles, 2 aligned b128 reads each
    union { f16x16 v; f16x8 h[2]; } afr[2];
#pragma unroll
    for (int mi = 0; mi < 2; ++mi) {
      afr[mi].h[0] = *(const f16x8*)(&sA[a_base + mi * 16 * LDP]);
      afr[mi].h[1] = *(const f16x8*)(&sA[a_base + mi * 16 * LDP + 16]);
    }

    // B fragments: 8 total (2 N-subtiles x 4 gates); each feeds both A frags
    union { f16x16 v; f16x8 h[2]; } bfr[2];
    bfr[0].h[0] = *(const f16x8*)(&sW[b_base]);
    bfr[0].h[1] = *(const f16x8*)(&sW[b_base + 16]);
#pragma unroll
    for (int f = 0; f < 8; ++f) {         // f = g*2 + ni
      if (f < 7) {
        const int fn = f + 1;
        const int bb = b_base + ((fn >> 1) * BN + (fn & 1) * 16) * LDP;
        bfr[(f + 1) & 1].h[0] = *(const f16x8*)(&sW[bb]);
        bfr[(f + 1) & 1].h[1] = *(const f16x8*)(&sW[bb + 16]);
      }
#pragma unroll
      for (int mi = 0; mi < 2; ++mi) {
        const int ai = f * 2 + mi;        // ((g*2+ni)*2)+mi
        acc[ai] = __builtin_amdgcn_wmma_f32_16x16x32_f16(
            false, afr[mi].v, false, bfr[f & 1].v, (short)0, acc[ai],
            false, false);
      }
    }
    p ^= 1;
  }

  // ---- fused LSTM epilogue ----
  // C/D layout: VGPR r holds M=r (lanes 0-15) / M=8+r (lanes 16-31), N=lane%16
  const int grp = lane >> 4;
#pragma unroll
  for (int ni = 0; ni < 2; ++ni) {
    const int col = blockN + wn * 32 + ni * 16 + nl;  // per-gate column
    const float bF = bias[0 * OUTCH + col];
    const float bI = bias[1 * OUTCH + col];
    const float bO = bias[2 * OUTCH + col];
    const float bS = bias[3 * OUTCH + col];
#pragma unroll
    for (int mi = 0; mi < 2; ++mi) {
      const f32x8 aF = acc[((0 * 2 + ni) * 2) + mi];
      const f32x8 aI = acc[((1 * 2 + ni) * 2) + mi];
      const f32x8 aO = acc[((2 * 2 + ni) * 2) + mi];
      const f32x8 aS = acc[((3 * 2 + ni) * 2) + mi];
#pragma unroll
      for (int r = 0; r < 8; ++r) {
        const int row = blockM + wm * 32 + mi * 16 + grp * 8 + r;
        const size_t idx = (size_t)row * OUTCH + col;
        const float fg = fast_sigmoid(aF[r] + bF);
        const float ig = fast_sigmoid(aI[r] + bI);
        const float og = fast_sigmoid(aO[r] + bO);
        const float sg = fast_sigmoid(aS[r] + bS);
        const float nc = fg * cell[idx] + ig * sg;
        const float nh = og * fast_tanh(nc);
        outH[idx] = nh;
        outC[idx] = nc;
      }
    }
  }
}

// ---------------- host side ----------------
extern "C" void kernel_launch(void* const* d_in, const int* in_sizes, int n_in,
                              void* d_out, int out_size, void* d_ws, size_t ws_size,
                              hipStream_t stream) {
  const float* input = (const float*)d_in[0];
  // d_in[1] = hidden: unused by the reference computation
  const float* cell  = (const float*)d_in[2];
  const float* W     = (const float*)d_in[3];
  const float* b     = (const float*)d_in[4];

  float* outH = (float*)d_out;
  float* outC = outH + (size_t)BATCH * OUTCH;

  // workspace: f16 input copy (32 MB) + f16 transposed W (32 MB)
  _Float16* Ah = (_Float16*)d_ws;
  _Float16* Wh = Ah + (size_t)BATCH * INCH;

  const int n4a = (BATCH * INCH) / 4;
  lstm_cvt_f32_f16<<<(n4a + 255) / 256, 256, 0, stream>>>(input, Ah, n4a);

  dim3 tgrid(WROW / 32, INCH / 32);
  lstm_cvt_w_transpose<<<tgrid, dim3(32, 8), 0, stream>>>(W, Wh);

  dim3 grid(OUTCH / BN, BATCH / BM);  // 32 x 64 blocks
  lstm_wmma_gemm<<<grid, 256, 0, stream>>>(Ah, Wh, b, cell, outH, outC);
}
